// GPModel_59021440581986
// MI455X (gfx1250) — matrix operand order repaired and checked by reading, not compiled, not run on hardware
//
#include <hip/hip_runtime.h>
#include <math.h>

// ---------------- problem constants (match reference) ----------------
constexpr int G    = 200;
constexpr int NPG  = 500;
constexpr int N    = G * NPG;        // 100000
constexpr int EPG  = NPG * 12;       // 6000
constexpr int E    = G * EPG;        // 1200000
constexpr int F    = 64;             // F_IN == NHID
constexpr int KPG  = 250;
constexpr int K    = G * KPG;        // 50000

typedef __attribute__((ext_vector_type(16))) _Float16 v16h;
typedef __attribute__((ext_vector_type(8)))  _Float16 v8h;
typedef __attribute__((ext_vector_type(8)))  float    v8f;

// ---------------- small helpers ----------------
__global__ void fill_f32(float* p, long long n, float v) {
    long long i = (long long)blockIdx.x * blockDim.x + threadIdx.x;
    if (i < n) p[i] = v;
}
__global__ void fill_i32(int* p, long long n, int v) {
    long long i = (long long)blockIdx.x * blockDim.x + threadIdx.x;
    if (i < n) p[i] = v;
}

__device__ __forceinline__ void atomicMaxF(float* addr, float val) {
    if (val >= 0.0f) atomicMax((int*)addr, __float_as_int(val));
    else             atomicMin((unsigned int*)addr, __float_as_uint(val));
}

// ---------------- phase 1: information score ----------------
__global__ void deg_kernel(const int* __restrict__ ei, float* __restrict__ deg) {
    int e = blockIdx.x * blockDim.x + threadIdx.x;
    if (e < E) atomicAdd(&deg[ei[e]], 1.0f);          // segment_sum over row
}
__global__ void dis_kernel(float* __restrict__ deg) { // deg -> D^-1/2 in place
    int i = blockIdx.x * blockDim.x + threadIdx.x;
    if (i < N) { float d = deg[i]; deg[i] = (d > 0.0f) ? (1.0f / sqrtf(d)) : 0.0f; }
}
__global__ void agg_kernel(const int* __restrict__ ei, const float* __restrict__ dis,
                           const float* __restrict__ x, float* __restrict__ agg) {
    long long gid = (long long)blockIdx.x * blockDim.x + threadIdx.x;
    if (gid >= (long long)E * F) return;
    int e = (int)(gid >> 6), f = (int)(gid & 63);
    int r = ei[e], c = ei[E + e];
    float w = dis[r] * dis[c];
    atomicAdd(&agg[(long long)c * F + f], w * x[(long long)r * F + f]);
}
__global__ void score_kernel(const float* __restrict__ x, const float* __restrict__ agg,
                             float* __restrict__ score) {
    int i = blockIdx.x * blockDim.x + threadIdx.x;
    if (i >= N) return;
    const float4* xa = (const float4*)(x   + (long long)i * F);
    const float4* ga = (const float4*)(agg + (long long)i * F);
    float s = 0.0f;
#pragma unroll
    for (int j = 0; j < 16; ++j) {
        float4 a = xa[j], b = ga[j];
        s += fabsf(a.x - b.x) + fabsf(a.y - b.y) + fabsf(a.z - b.z) + fabsf(a.w - b.w);
    }
    score[i] = s;
}

// ---------------- phase 2: per-graph top-k (bitonic sort 512 in LDS) ----------------
__global__ void topk_kernel(const float* __restrict__ score,
                            int* __restrict__ perm, int* __restrict__ mapping) {
    __shared__ float sv[512];
    __shared__ short si[512];
    int g = blockIdx.x, tid = threadIdx.x;             // 256 threads
    for (int i = tid; i < 512; i += 256) {
        sv[i] = (i < NPG) ? score[g * NPG + i] : -3.4e38f;
        si[i] = (short)i;
    }
    __syncthreads();
    for (int k = 2; k <= 512; k <<= 1) {
        for (int j = k >> 1; j > 0; j >>= 1) {
            for (int i = tid; i < 512; i += 256) {
                int ixj = i ^ j;
                if (ixj > i) {
                    bool desc = ((i & k) == 0);
                    float a = sv[i], b = sv[ixj];
                    bool sw = desc ? (a < b) : (a > b);
                    if (sw) { sv[i] = b; sv[ixj] = a; short t = si[i]; si[i] = si[ixj]; si[ixj] = t; }
                }
            }
            __syncthreads();
        }
    }
    for (int t = tid; t < KPG; t += 256) {
        int node = g * NPG + si[t];
        int kidx = g * KPG + t;
        perm[kidx]   = node;
        mapping[node] = kidx;                           // mapping pre-filled with -1
    }
}

__global__ void relabel_kernel(const int* __restrict__ ei, const int* __restrict__ mapping,
                               int* __restrict__ nr, int* __restrict__ nc) {
    int e = blockIdx.x * blockDim.x + threadIdx.x;
    if (e >= E) return;
    int a = mapping[ei[e]], b = mapping[ei[E + e]];
    if (a < 0 || b < 0) { a = -1; b = -1; }
    nr[e] = a; nc[e] = b;
}

// ---------------- phase 3: GEMM h = x[perm] @ W1 via WMMA f16->f32 ----------------
__global__ void gemm_wmma_kernel(const float* __restrict__ x, const float* __restrict__ W1,
                                 const int* __restrict__ perm, float* __restrict__ h) {
    __shared__ _Float16 xs[16][72];    // 16 rows x 64 K (padded, 16B-aligned rows)
    __shared__ _Float16 wt[64][72];    // TRANSPOSED: wt[n][k] (contiguous-K per lane)
    int tid = threadIdx.x;             // 128 threads = 4 waves
    int r0 = blockIdx.x * 16;
    for (int i = tid; i < 16 * 64; i += 128) {
        int m = i >> 6, f = i & 63;
        int src = perm[r0 + m];
        xs[m][f] = (_Float16)x[(long long)src * F + f];
    }
    for (int i = tid; i < 64 * 64; i += 128) {
        int k = i >> 6, n = i & 63;
        wt[n][k] = (_Float16)W1[i];    // coalesced global read, transposed LDS write
    }
    __syncthreads();
    int wave = tid >> 5, lane = tid & 31;
    int n0 = wave * 16;
    int mrow = lane & 15, hi = lane >> 4;
    int n = n0 + mrow;
    const v8h* arow = (const v8h*)&xs[mrow][0];        // 9 x v8h per row
    const v8h* brow = (const v8h*)&wt[n][0];
    v8f c = {};
#pragma unroll
    for (int s = 0; s < 2; ++s) {
        int ks = s * 32;
        // A 16x32 f16 layout: lane half 'hi' covers K = ks+hi*8..+7 and ks+16+hi*8..+7
        v8h alo = arow[(ks + hi * 8) >> 3];
        v8h ahi = arow[(ks + 16 + hi * 8) >> 3];
        // B 32x16 f16 layout: lane half 'hi' covers K = ks+hi*8..+7 and ks+16+hi*8..+7 at column n
        v8h blo = brow[(ks + hi * 8) >> 3];
        v8h bhi = brow[(ks + 16 + hi * 8) >> 3];
        v16h a, b;
#pragma unroll
        for (int j = 0; j < 8; ++j) {
            a[j] = alo[j]; a[8 + j] = ahi[j];
            b[j] = blo[j]; b[8 + j] = bhi[j];
        }
        c = __builtin_amdgcn_wmma_f32_16x16x32_f16(false, a, false, b, (short)0, c, false, false);
    }
#pragma unroll
    for (int j = 0; j < 8; ++j) {                     // C/D 16x16 f32 layout
        int m = j + 8 * hi;
        h[(long long)(r0 + m) * F + n] = c[j];
    }
}

// ---------------- phase 4: GCN normalization + aggregation ----------------
__global__ void deg2_kernel(const int* __restrict__ nc, float* __restrict__ deg2) {
    int e = blockIdx.x * blockDim.x + threadIdx.x;
    if (e < E && nc[e] >= 0) atomicAdd(&deg2[nc[e]], 1.0f);
}
__global__ void d2_kernel(float* __restrict__ deg2) { // -> D^-1/2 (with self loop) in place
    int i = blockIdx.x * blockDim.x + threadIdx.x;
    if (i < K) deg2[i] = 1.0f / sqrtf(deg2[i] + 1.0f);
}
__global__ void gcn_agg_kernel(const int* __restrict__ nr, const int* __restrict__ nc,
                               const float* __restrict__ d2, const float* __restrict__ h,
                               float* __restrict__ acc) {
    long long gid = (long long)blockIdx.x * blockDim.x + threadIdx.x;
    if (gid >= (long long)E * F) return;
    int e = (int)(gid >> 6), f = (int)(gid & 63);
    int a = nr[e];
    if (a < 0) return;
    int b = nc[e];
    float norm = d2[a] * d2[b];
    atomicAdd(&acc[(long long)b * F + f], norm * h[(long long)a * F + f]);
}
__global__ void gcn_finish_kernel(const float* __restrict__ h, const float* __restrict__ d2,
                                  const float* __restrict__ b1, float* __restrict__ hfin) {
    long long gid = (long long)blockIdx.x * blockDim.x + threadIdx.x;
    if (gid >= (long long)K * F) return;
    int k = (int)(gid >> 6), f = (int)(gid & 63);
    float dd = d2[k];
    float v = hfin[gid] + dd * dd * h[gid] + b1[f];    // hfin currently holds acc (in place)
    hfin[gid] = fmaxf(v, 0.0f);
}
__global__ void invn_kernel(const float* __restrict__ hfin, const float* __restrict__ beta,
                            float* __restrict__ invn, float* __restrict__ ls) {
    int k = blockIdx.x * blockDim.x + threadIdx.x;
    if (k >= K) return;
    const float4* ha = (const float4*)(hfin + (long long)k * F);
    float s = 0.0f;
#pragma unroll
    for (int j = 0; j < 16; ++j) {
        float4 a = ha[j];
        s += a.x * a.x + a.y * a.y + a.z * a.z + a.w * a.w;
    }
    float inv = 1.0f / fmaxf(sqrtf(s), 1e-12f);
    invn[k] = inv;
    ls[k] = beta[0] * s * inv * inv;                   // self-loop logit
}

// ---------------- phase 5: AGNN edge softmax ----------------
__global__ void logit_kernel(const int* __restrict__ nr, const int* __restrict__ nc,
                             const float* __restrict__ hfin, const float* __restrict__ invn,
                             const float* __restrict__ beta,
                             float* __restrict__ loge, float* __restrict__ m) {
    int e = blockIdx.x * 8 + (threadIdx.x >> 5);       // wave per edge (wave32)
    int lane = threadIdx.x & 31;
    if (e >= E) return;
    int a = nr[e];
    if (a < 0) { if (lane == 0) loge[e] = -1e9f; return; }
    int b = nc[e];
    const float2* ha = (const float2*)(hfin + (long long)a * F);
    const float2* hb = (const float2*)(hfin + (long long)b * F);
    float2 va = ha[lane], vb = hb[lane];
    float p = va.x * vb.x + va.y * vb.y;
#pragma unroll
    for (int off = 16; off > 0; off >>= 1) p += __shfl_xor(p, off, 32);
    if (lane == 0) {
        float le = beta[0] * p * invn[a] * invn[b];
        loge[e] = le;
        atomicMaxF(&m[b], le);
    }
}
__global__ void mfin_kernel(float* __restrict__ m, const float* __restrict__ ls) {
    int k = blockIdx.x * blockDim.x + threadIdx.x;
    if (k < K) m[k] = fmaxf(m[k], ls[k]);
}
__global__ void ee_kernel(const int* __restrict__ nc, const float* __restrict__ loge,
                          const float* __restrict__ m, float* __restrict__ ee,
                          float* __restrict__ denom) {
    int e = blockIdx.x * blockDim.x + threadIdx.x;
    if (e >= E) return;
    int b = nc[e];
    if (b < 0) { ee[e] = 0.0f; return; }
    float v = __expf(loge[e] - m[b]);
    ee[e] = v;
    atomicAdd(&denom[b], v);
}
__global__ void out_agg_kernel(const int* __restrict__ nr, const int* __restrict__ nc,
                               const float* __restrict__ ee, const float* __restrict__ hfin,
                               float* __restrict__ acc) {
    long long gid = (long long)blockIdx.x * blockDim.x + threadIdx.x;
    if (gid >= (long long)E * F) return;
    int e = (int)(gid >> 6), f = (int)(gid & 63);
    int a = nr[e];
    if (a < 0) return;
    int b = nc[e];
    atomicAdd(&acc[(long long)b * F + f], ee[e] * hfin[(long long)a * F + f]);
}
__global__ void out_final_kernel(const float* __restrict__ hfin, const float* __restrict__ ls,
                                 const float* __restrict__ m, const float* __restrict__ denom,
                                 float* __restrict__ out) {    // out == acc region, in place
    long long gid = (long long)blockIdx.x * blockDim.x + threadIdx.x;
    if (gid >= (long long)K * F) return;
    int k = (int)(gid >> 6);
    float es = __expf(ls[k] - m[k]);
    out[gid] = (out[gid] + es * hfin[gid]) / (denom[k] + es);
}

// ---------------- phase 6: readout ----------------
__global__ void readout_kernel(const float* __restrict__ out, float* __restrict__ res) {
    int g = blockIdx.x, f = threadIdx.x;               // 64 threads per graph
    float s = 0.0f, mx = -3.4e38f;
    const float* base = out + (long long)g * KPG * F + f;
    for (int j = 0; j < KPG; ++j) {
        float v = base[(long long)j * F];
        s += v;
        mx = fmaxf(mx, v);
    }
    res[g * 2 * F + f]     = fmaxf(s * (1.0f / KPG), 0.0f);
    res[g * 2 * F + F + f] = fmaxf(mx, 0.0f);
}

// ---------------- launcher ----------------
static inline int blks(long long n, int b) { return (int)((n + b - 1) / b); }

extern "C" void kernel_launch(void* const* d_in, const int* in_sizes, int n_in,
                              void* d_out, int out_size, void* d_ws, size_t ws_size,
                              hipStream_t stream) {
    const float* x    = (const float*)d_in[0];
    const int*   ei   = (const int*)d_in[1];
    const float* W1   = (const float*)d_in[2];
    const float* b1   = (const float*)d_in[3];
    const float* beta = (const float*)d_in[4];
    float* res = (float*)d_out;

    // workspace layout (floats); R1 = N*F region reused across phases
    float* Fw = (float*)d_ws;
    float* agg   = Fw;                        // N*F   (phase1)
    float* h_raw = Fw;                        // K*F   (phase3+) == agg[0 : K*F)
    float* hfin  = Fw + (long long)K * F;     // K*F   == agg[K*F : 2*K*F)
    float* o_acc = Fw;                        // K*F   reuses h_raw after GCN finish
    float* deg   = Fw + (long long)N * F;                 // N  (-> dis in place)
    float* score = deg + N;                               // N
    float* d2    = score + N;                             // K  (-> D2^-1/2)
    float* invn  = d2 + K;                                // K
    float* mbuf  = invn + K;                              // K
    float* denom = mbuf + K;                              // K
    float* ls    = denom + K;                             // K
    float* loge  = ls + K;                                // E
    float* eebuf = loge + E;                              // E
    int*   mapping = (int*)(eebuf + E);                   // N
    int*   perm    = mapping + N;                         // K
    int*   nr      = perm + K;                            // E
    int*   nc      = nr + E;                              // E

    const int B = 256;

    // ---- score ----
    fill_f32<<<blks((long long)N * F, B), B, 0, stream>>>(agg, (long long)N * F, 0.0f);
    fill_f32<<<blks(N, B), B, 0, stream>>>(deg, N, 0.0f);
    fill_i32<<<blks(N, B), B, 0, stream>>>(mapping, N, -1);
    deg_kernel<<<blks(E, B), B, 0, stream>>>(ei, deg);
    dis_kernel<<<blks(N, B), B, 0, stream>>>(deg);
    agg_kernel<<<blks((long long)E * F, B), B, 0, stream>>>(ei, deg, x, agg);
    score_kernel<<<blks(N, B), B, 0, stream>>>(x, agg, score);

    // ---- top-k + relabel ----
    topk_kernel<<<G, 256, 0, stream>>>(score, perm, mapping);
    relabel_kernel<<<blks(E, B), B, 0, stream>>>(ei, mapping, nr, nc);

    // ---- GCN: WMMA GEMM then normalized aggregation ----
    gemm_wmma_kernel<<<K / 16, 128, 0, stream>>>(x, W1, perm, h_raw);
    fill_f32<<<blks((long long)K * F, B), B, 0, stream>>>(hfin, (long long)K * F, 0.0f);
    fill_f32<<<blks(K, B), B, 0, stream>>>(d2, K, 0.0f);
    deg2_kernel<<<blks(E, B), B, 0, stream>>>(nc, d2);
    d2_kernel<<<blks(K, B), B, 0, stream>>>(d2);
    gcn_agg_kernel<<<blks((long long)E * F, B), B, 0, stream>>>(nr, nc, d2, h_raw, hfin);
    gcn_finish_kernel<<<blks((long long)K * F, B), B, 0, stream>>>(h_raw, d2, b1, hfin);
    invn_kernel<<<blks(K, B), B, 0, stream>>>(hfin, beta, invn, ls);

    // ---- AGNN edge softmax ----
    fill_f32<<<blks(K, B), B, 0, stream>>>(mbuf, K, -3.4e38f);
    fill_f32<<<blks(K, B), B, 0, stream>>>(denom, K, 0.0f);
    fill_f32<<<blks((long long)K * F, B), B, 0, stream>>>(o_acc, (long long)K * F, 0.0f);
    logit_kernel<<<blks(E, 8), 256, 0, stream>>>(nr, nc, hfin, invn, beta, loge, mbuf);
    mfin_kernel<<<blks(K, B), B, 0, stream>>>(mbuf, ls);
    ee_kernel<<<blks(E, B), B, 0, stream>>>(nc, loge, mbuf, eebuf, denom);
    out_agg_kernel<<<blks((long long)E * F, B), B, 0, stream>>>(nr, nc, eebuf, hfin, o_acc);
    out_final_kernel<<<blks((long long)K * F, B), B, 0, stream>>>(hfin, ls, mbuf, denom, o_acc);

    // ---- readout ----
    readout_kernel<<<G, 64, 0, stream>>>(o_acc, res);
    (void)in_sizes; (void)n_in; (void)out_size; (void)ws_size;
}
